// SingleHeadAttention_12506944766493
// MI455X (gfx1250) — matrix operand order repaired and checked by reading, not compiled, not run on hardware
//
#include <hip/hip_runtime.h>

// Problem constants (match reference)
#define BB   4
#define TT   4096
#define CC   1024
#define NQKV (3 * CC)
#define SCALE 0.03125f   // C^-0.5 = 1/32

typedef __bf16 bf16;
typedef __attribute__((ext_vector_type(16))) __bf16 v16bf;
typedef __attribute__((ext_vector_type(8)))  float  v8f;
typedef unsigned int       u32x4 __attribute__((ext_vector_type(4)));
typedef int                i32x4 __attribute__((ext_vector_type(4)));
typedef int                i32x8 __attribute__((ext_vector_type(8)));

// ---- CDNA5 Tensor Data Mover (probe-verified 5-arg builtin on ROCm 7.2) ----
#if defined(__has_builtin)
#if __has_builtin(__builtin_amdgcn_tensor_load_to_lds) && \
    __has_builtin(__builtin_amdgcn_s_wait_tensorcnt)
#define USE_TDM 1
#endif
#endif

#if USE_TDM
// Fire a 2-D TDM tile load: tile_d0 bf16 elems/row, tile_d1 rows, global row
// stride `stride_elems`; D# pad fields insert LDS padding every
// (1<<pad_interval_enum) qwords, producing a bank-skewed LDS leading dim
// directly from the DMA engine. Wave-level op (EXEC ignored); tracked with
// TENSORcnt.
__device__ __forceinline__ void tdm_load_2d(unsigned int lds_addr,
                                            const void* gsrc,
                                            unsigned int tile_d0,
                                            unsigned int tile_d1,
                                            unsigned int stride_elems,
                                            unsigned int pad_interval_enum,
                                            unsigned int pad_amount_enum) {
  const unsigned long long ga = (unsigned long long)(size_t)gsrc;
  u32x4 g0;
  g0[0] = 1u;                                        // count=1 valid descriptor
  g0[1] = lds_addr;                                  // LDS byte address
  g0[2] = (unsigned int)ga;                          // global_addr[31:0]
  g0[3] = (unsigned int)(ga >> 32) | 0x80000000u;    // addr[56:32] | type=2
  i32x8 g1;
  g1[0] = (int)((1u << 16) |                         // data_size = 2B
                (1u << 20) |                         // pad_enable
                (pad_interval_enum << 22) | (pad_amount_enum << 25));
  g1[1] = (int)((stride_elems & 0xffffu) << 16);     // tensor_dim0 lo16
  g1[2] = (int)(((stride_elems >> 16) & 0xffffu) | ((unsigned)TT << 16));
  g1[3] = (int)((((unsigned)TT >> 16) & 0xffffu) | (tile_d0 << 16));
  g1[4] = (int)(tile_d1 & 0xffffu);                  // tile_dim1 (tile_dim2=0)
  g1[5] = (int)stride_elems;                         // tensor_dim0_stride lo32
  g1[6] = 0;
  g1[7] = 0;
  i32x4 z4 = {0, 0, 0, 0};
#if defined(__clang_major__) && (__clang_major__ >= 23)
  i32x8 z8 = {0, 0, 0, 0, 0, 0, 0, 0};
  __builtin_amdgcn_tensor_load_to_lds(g0, g1, z4, z4, z8, 0);
#else
  __builtin_amdgcn_tensor_load_to_lds(g0, g1, z4, z4, 0);
#endif
}
#endif

// ---------------------------------------------------------------------------
// WMMA helpers (gfx1250, wave32). D = A(16x32 bf16) x B(32x16 bf16) + C(f32)
// ---------------------------------------------------------------------------
__device__ __forceinline__ v8f wmma_bf16(v16bf a, v16bf b, v8f c) {
  return __builtin_amdgcn_wmma_f32_16x16x32_bf16(
      /*neg_a=*/false, a, /*neg_b=*/false, b,
      /*c_mod=*/(short)0, c, /*reuse_a=*/false, /*reuse_b=*/false);
}

// A fragment: 16x32 (MxK) bf16, row-major source, leading dim `ld`.
// k order used here (not the ISA table): lanes 0-15 -> M=lane, K in {0..15}; lanes 16-31 -> K in {16..31}.
__device__ __forceinline__ v16bf load_frag_a(const bf16* A, int ld) {
  const int lane = threadIdx.x & 31;
  const int row  = lane & 15;
  const int ko   = (lane < 16) ? 0 : 16;
  const bf16* r = A + row * ld;
  v16bf a;
#pragma unroll
  for (int i = 0; i < 8; ++i) {
    const int k = 2 * i + ko;
    a[2 * i]     = r[k];
    a[2 * i + 1] = r[k + 1];
  }
  return a;
}

// B fragment 32x16 (KxN), source stored N-major: src[n*ld + k] (x*W^T, Q*K^T).
__device__ __forceinline__ v16bf load_frag_b_nk(const bf16* src, int ld) {
  const int lane = threadIdx.x & 31;
  const int n    = lane & 15;
  const int k0   = (lane < 16) ? 0 : 16;
  const bf16* r = src + n * ld + k0;
  v16bf b;
#pragma unroll
  for (int e = 0; e < 16; ++e) b[e] = r[e];
  return b;
}

// B fragment 32x16 (KxN), source stored K-major: src[k*ld + n] (P*V).
__device__ __forceinline__ v16bf load_frag_b_kn(const bf16* src, int ld) {
  const int lane = threadIdx.x & 31;
  const int n    = lane & 15;
  const int k0   = (lane < 16) ? 0 : 16;
  v16bf b;
#pragma unroll
  for (int e = 0; e < 16; ++e) b[e] = src[(k0 + e) * ld + n];
  return b;
}

// f32 C/D 16x16 tile <-> row-major memory.
__device__ __forceinline__ v8f load_tile_c(const float* C, int ld) {
  const int lane = threadIdx.x & 31;
  const int col  = lane & 15;
  const int rb   = (lane < 16) ? 0 : 8;
  v8f v;
#pragma unroll
  for (int r = 0; r < 8; ++r) v[r] = C[(rb + r) * ld + col];
  return v;
}
__device__ __forceinline__ void store_tile_c(float* C, int ld, v8f v) {
  const int lane = threadIdx.x & 31;
  const int col  = lane & 15;
  const int rb   = (lane < 16) ? 0 : 8;
#pragma unroll
  for (int r = 0; r < 8; ++r) C[(rb + r) * ld + col] = v[r];
}

// ---------------------------------------------------------------------------
// Kernel 1: qkv = x @ W_qkv^T   (f32 in, bf16 out). Both tiles need f32->bf16
// conversion so staging stays on the VALU path.
// ---------------------------------------------------------------------------
#define GK 32
#define GLDA 34

__global__ __launch_bounds__(128) void qkv_gemm_kernel(
    const float* __restrict__ X, const float* __restrict__ W,
    bf16* __restrict__ Y, int M, int N, int K) {
  __shared__ bf16 As[64][GLDA];
  __shared__ bf16 Bs[64][GLDA];
  const int m0 = blockIdx.y * 64;
  const int n0 = blockIdx.x * 64;
  const int t  = threadIdx.x;
  const int wv = t >> 5;
  const int wrow = wv * 16;

  v8f acc[4] = {};
  for (int k0 = 0; k0 < K; k0 += GK) {
    {
      const int row = t >> 1;
      const int cb  = (t & 1) * 16;
      const float* sa = X + (size_t)(m0 + row) * K + k0 + cb;
      const float* sb = W + (size_t)(n0 + row) * K + k0 + cb;
      if (k0 + GK < K) {
        __builtin_prefetch(sa + GK, 0, 1);
        __builtin_prefetch(sb + GK, 0, 1);
      }
#pragma unroll
      for (int e = 0; e < 16; e += 4) {
        float4 fa = *(const float4*)(sa + e);
        float4 fb = *(const float4*)(sb + e);
        As[row][cb + e + 0] = (bf16)fa.x; As[row][cb + e + 1] = (bf16)fa.y;
        As[row][cb + e + 2] = (bf16)fa.z; As[row][cb + e + 3] = (bf16)fa.w;
        Bs[row][cb + e + 0] = (bf16)fb.x; Bs[row][cb + e + 1] = (bf16)fb.y;
        Bs[row][cb + e + 2] = (bf16)fb.z; Bs[row][cb + e + 3] = (bf16)fb.w;
      }
    }
    __syncthreads();
    v16bf a = load_frag_a(&As[wrow][0], GLDA);
#pragma unroll
    for (int j = 0; j < 4; ++j) {
      v16bf b = load_frag_b_nk(&Bs[j * 16][0], GLDA);
      acc[j] = wmma_bf16(a, b, acc[j]);
    }
    __syncthreads();
  }
  const int lane = t & 31;
  const int col  = lane & 15;
  const int rb   = (lane < 16) ? 0 : 8;
#pragma unroll
  for (int j = 0; j < 4; ++j)
#pragma unroll
    for (int r = 0; r < 8; ++r)
      Y[(size_t)(m0 + wrow + rb + r) * N + n0 + j * 16 + col] = (bf16)acc[j][r];
}

// ---------------------------------------------------------------------------
// Kernel 2: causal flash attention. Q tile (32x1024 bf16) + O accumulator
// (32x1024 f32) resident in the 320KB WGP LDS. Q, K, V all staged by TDM
// (tensor_load_to_lds + s_wait_tensorcnt) when available, else ds stores.
//
// Q TDM layout: pad 16B after every 1024B of data -> each 2048B row becomes
// two 1040B segments (row stride 2080B = QLD). S-loop chunks (32 elems) never
// straddle the 512-element boundary, so addressing adds +8 elems for kd>=512.
// ---------------------------------------------------------------------------
#define BQ 32
#define BK 64
#define QLD 1040  // bf16 row stride (2080B)
#define QSHIFT(c) ((c) >= 512 ? (c) + 8 : (c))
#define OLD 1032  // f32 : row = 4128B
#define KLD 40    // bf16: row = 80B  (64B data + 16B pad)
#define VLD 72    // bf16: row = 144B (128B data + 16B pad)
#define PLD 66

#define SM_Q  (BQ * QLD * 2)            // 66560
#define SM_O  (BQ * OLD * 4)            // 132096
#define SM_K  (BK * KLD * 2)            // 5120
#define SM_V  (BK * VLD * 2)            // 9216
#define SM_P  (BQ * PLD * 2)            // 4224
#define SM_FL ((3 * BQ + 4 * BQ) * 4)   // 896
#define SM_TOTAL (SM_Q + SM_O + SM_K + SM_V + SM_P + SM_FL)  // 218,112 B

__global__ __launch_bounds__(128) void flash_attn_kernel(
    const bf16* __restrict__ QKV, bf16* __restrict__ Oout) {
  extern __shared__ char smem[];
  bf16*  Qs      = (bf16*)smem;
  float* O       = (float*)(smem + SM_Q);
  bf16*  Ks      = (bf16*)(smem + SM_Q + SM_O);
  bf16*  Vs      = Ks + BK * KLD;
  bf16*  Ps      = Vs + BK * VLD;
  float* row_m   = (float*)(smem + SM_Q + SM_O + SM_K + SM_V + SM_P);
  float* row_l   = row_m + BQ;
  float* row_al  = row_l + BQ;
  float* pm_part = row_al + BQ;   // [2][BQ]
  float* pl_part = pm_part + 2 * BQ;

  const int b    = blockIdx.y;
  const int qb0  = blockIdx.x * BQ;
  const int t    = threadIdx.x;
  const int lane = t & 31;
  const int wv   = t >> 5;

  const bf16* Qg = QKV + (size_t)(b * TT + qb0) * NQKV;   // q cols [0,C)
  const bf16* Kg = QKV + (size_t)b * TT * NQKV + CC;      // k cols [C,2C)
  const bf16* Vg = QKV + (size_t)b * TT * NQKV + 2 * CC;  // v cols [2C,3C)

  for (int idx = t; idx < BQ * OLD; idx += 128) O[idx] = 0.f;
  if (t < BQ) { row_m[t] = -1e30f; row_l[t] = 0.f; }

  // ---- stage Q tile ----
#if USE_TDM
  if (wv == 0) {
    // 32 rows x 1024 bf16; pad 4 dwords every 128 qwords (enum 7 = 1024B)
    tdm_load_2d((unsigned int)(size_t)Qs, Qg, CC, BQ, NQKV, 7, 3);
    __builtin_amdgcn_s_wait_tensorcnt(0);
  }
#else
  for (int idx = t; idx < BQ * (CC / 8); idx += 128) {
    const int row = idx >> 7;
    const int c8  = (idx & 127) * 8;
    *(uint4*)&Qs[row * QLD + QSHIFT(c8)] =
        *(const uint4*)(Qg + (size_t)row * NQKV + c8);
  }
#endif
  __syncthreads();

  // Wave roles for S: rows (wv&1)*16..+15, cols (wv>>1)*32..+31
  const int srow0 = (wv & 1) * 16;
  const int sc0   = (wv >> 1) * 32;
  const int ch    = wv >> 1;
  const int col   = lane & 15;
  const int rb    = (lane < 16) ? 0 : 8;

  const int jmax = (qb0 + BQ - 1) / BK;   // causal: skip fully-masked blocks
  for (int j = 0; j <= jmax; ++j) {
    const int kb0 = j * BK;

    // ---- S = Q * K^T over the 1024 inner dim, 32 at a time ----
    v8f s0 = {}, s1 = {};
    for (int kd = 0; kd < CC; kd += 32) {
#if USE_TDM
      if (wv == 0) {
        // 64 rows x 32 bf16; pad 4 dwords every 8 qwords (enum 3) -> KLD=40
        tdm_load_2d((unsigned int)(size_t)Ks,
                    Kg + (size_t)kb0 * NQKV + kd, 32, 64, NQKV, 3, 3);
        __builtin_amdgcn_s_wait_tensorcnt(0);
      }
#else
      {
        const int row = t >> 1;
        const int cb  = (t & 1) * 16;
        const bf16* src = Kg + (size_t)(kb0 + row) * NQKV + kd + cb;
#pragma unroll
        for (int e = 0; e < 16; ++e) Ks[row * KLD + cb + e] = src[e];
      }
#endif
      __syncthreads();
      v16bf a = load_frag_a(Qs + srow0 * QLD + QSHIFT(kd), QLD);
      s0 = wmma_bf16(a, load_frag_b_nk(Ks + sc0 * KLD, KLD), s0);
      s1 = wmma_bf16(a, load_frag_b_nk(Ks + (sc0 + 16) * KLD, KLD), s1);
      __syncthreads();
    }

    // ---- scale, causal mask, per-wave row-max (16-lane butterfly) ----
    float sv0[8], sv1[8], pm[8];
#pragma unroll
    for (int r = 0; r < 8; ++r) {
      const int qpos = qb0 + srow0 + rb + r;
      const int kp0  = kb0 + sc0 + col;
      sv0[r] = (kp0      <= qpos) ? s0[r] * SCALE : -1e30f;
      sv1[r] = (kp0 + 16 <= qpos) ? s1[r] * SCALE : -1e30f;
      float m = fmaxf(sv0[r], sv1[r]);
#pragma unroll
      for (int off = 8; off >= 1; off >>= 1) m = fmaxf(m, __shfl_xor(m, off));
      pm[r] = m;
    }
    if ((lane & 15) == 0) {
#pragma unroll
      for (int r = 0; r < 8; ++r) pm_part[ch * BQ + srow0 + rb + r] = pm[r];
    }
    __syncthreads();

    // ---- online softmax state update (one thread per row) ----
    if (t < BQ) {
      const float mo = row_m[t];
      const float mn = fmaxf(mo, fmaxf(pm_part[t], pm_part[BQ + t]));
      row_m[t]  = mn;
      row_al[t] = __expf(mo - mn);
    }
    __syncthreads();

    // ---- P = exp(S - m), row-sum partials, rescale O ----
    float ps[8];
#pragma unroll
    for (int r = 0; r < 8; ++r) {
      const int row = srow0 + rb + r;
      const float mn = row_m[row];
      const float p0 = __expf(sv0[r] - mn);
      const float p1 = __expf(sv1[r] - mn);
      Ps[row * PLD + sc0 + col]      = (bf16)p0;
      Ps[row * PLD + sc0 + 16 + col] = (bf16)p1;
      float s = p0 + p1;
#pragma unroll
      for (int off = 8; off >= 1; off >>= 1) s += __shfl_xor(s, off);
      ps[r] = s;
    }
    if ((lane & 15) == 0) {
#pragma unroll
      for (int r = 0; r < 8; ++r) pl_part[ch * BQ + srow0 + rb + r] = ps[r];
    }
    for (int idx = t; idx < BQ * CC; idx += 128) {
      const int row = idx >> 10;
      const int c   = idx & (CC - 1);
      O[row * OLD + c] *= row_al[row];
    }
    __syncthreads();
    if (t < BQ) row_l[t] = row_l[t] * row_al[t] + pl_part[t] + pl_part[BQ + t];

    // ---- O += P * V (stream V in 64x64 dim chunks) ----
    const int prt = (wv & 1) * 16;
    const int pdb = (wv >> 1) * 32;
    v16bf pa0 = load_frag_a(Ps + prt * PLD, PLD);       // keys 0..31
    v16bf pa1 = load_frag_a(Ps + prt * PLD + 32, PLD);  // keys 32..63
    for (int dc = 0; dc < CC; dc += 64) {
#if USE_TDM
      if (wv == 0) {
        // 64 rows x 64 bf16; pad 4 dwords every 16 qwords (enum 4) -> VLD=72
        tdm_load_2d((unsigned int)(size_t)Vs,
                    Vg + (size_t)kb0 * NQKV + dc, 64, 64, NQKV, 4, 3);
        __builtin_amdgcn_s_wait_tensorcnt(0);
      }
#else
      {
        const int key = t >> 1;
        const int cb  = (t & 1) * 32;
        const bf16* src = Vg + (size_t)(kb0 + key) * NQKV + dc + cb;
#pragma unroll
        for (int e = 0; e < 32; ++e) Vs[key * VLD + cb + e] = src[e];
      }
#endif
      __syncthreads();
#pragma unroll
      for (int dt = 0; dt < 2; ++dt) {
        float* Op = O + prt * OLD + dc + pdb + dt * 16;
        v8f c = load_tile_c(Op, OLD);
        c = wmma_bf16(pa0, load_frag_b_kn(Vs + pdb + dt * 16, VLD), c);
        c = wmma_bf16(pa1, load_frag_b_kn(Vs + 32 * VLD + pdb + dt * 16, VLD), c);
        store_tile_c(Op, OLD, c);
      }
      __syncthreads();
    }
  }

  // ---- normalize and emit O as bf16 ----
  __syncthreads();
  for (int idx = t; idx < BQ * CC; idx += 128) {
    const int row = idx >> 10;
    const int c   = idx & (CC - 1);
    const float inv = 1.0f / row_l[row];
    Oout[(size_t)(b * TT + qb0 + row) * CC + c] = (bf16)(O[row * OLD + c] * inv);
  }
}

// ---------------------------------------------------------------------------
// Kernel 3: out = o @ W_out^T (bf16 A TDM-staged; f32 W converted; f32 out)
// ---------------------------------------------------------------------------
#define ALD 40  // bf16 A tile leading dim (matches TDM pad: 64B data + 16B)

__global__ __launch_bounds__(128) void out_gemm_kernel(
    const bf16* __restrict__ Xb, const float* __restrict__ W,
    float* __restrict__ Y, int M, int N, int K) {
  __shared__ __align__(16) bf16 As[64][ALD];
  __shared__ bf16 Bs[64][GLDA];
  const int m0 = blockIdx.y * 64;
  const int n0 = blockIdx.x * 64;
  const int t  = threadIdx.x;
  const int wv = t >> 5;
  const int wrow = wv * 16;

  v8f acc[4] = {};
  for (int k0 = 0; k0 < K; k0 += GK) {
    // A tile: 64x32 bf16 via TDM (wave 0), B tile: f32->bf16 on VALU
#if USE_TDM
    if (wv == 0) {
      tdm_load_2d((unsigned int)(size_t)&As[0][0],
                  Xb + (size_t)m0 * K + k0, 32, 64, (unsigned)K, 3, 3);
    }
#else
    {
      const int row = t >> 1;
      const int cb  = (t & 1) * 16;
      const bf16* sa = Xb + (size_t)(m0 + row) * K + k0 + cb;
#pragma unroll
      for (int e = 0; e < 16; ++e) As[row][cb + e] = sa[e];
    }
#endif
    {
      const int row = t >> 1;
      const int cb  = (t & 1) * 16;
      const float* sb = W + (size_t)(n0 + row) * K + k0 + cb;
#pragma unroll
      for (int e = 0; e < 16; e += 4) {
        float4 fb = *(const float4*)(sb + e);
        Bs[row][cb + e + 0] = (bf16)fb.x; Bs[row][cb + e + 1] = (bf16)fb.y;
        Bs[row][cb + e + 2] = (bf16)fb.z; Bs[row][cb + e + 3] = (bf16)fb.w;
      }
    }
#if USE_TDM
    if (wv == 0) __builtin_amdgcn_s_wait_tensorcnt(0);
#endif
    __syncthreads();
    v16bf a = load_frag_a(&As[wrow][0], ALD);
#pragma unroll
    for (int j = 0; j < 4; ++j) {
      v16bf b = load_frag_b_nk(&Bs[j * 16][0], GLDA);
      acc[j] = wmma_bf16(a, b, acc[j]);
    }
    __syncthreads();
  }
  const int lane = t & 31;
  const int col  = lane & 15;
  const int rbs  = (lane < 16) ? 0 : 8;
#pragma unroll
  for (int j = 0; j < 4; ++j)
#pragma unroll
    for (int r = 0; r < 8; ++r)
      Y[(size_t)(m0 + wrow + rbs + r) * N + n0 + j * 16 + col] = acc[j][r];
}

// ---------------------------------------------------------------------------
// Launch: d_in = {x f32, mask bool(unused; causal recomputed), w_qkv, w_out}
// Workspace: qkv bf16 (96MB) | o bf16 (32MB)
// ---------------------------------------------------------------------------
extern "C" void kernel_launch(void* const* d_in, const int* in_sizes, int n_in,
                              void* d_out, int out_size, void* d_ws, size_t ws_size,
                              hipStream_t stream) {
  (void)in_sizes; (void)n_in; (void)out_size; (void)ws_size;
  const float* x     = (const float*)d_in[0];
  const float* w_qkv = (const float*)d_in[2];
  const float* w_out = (const float*)d_in[3];
  float* out = (float*)d_out;

  bf16* qkv = (bf16*)d_ws;
  bf16* o   = qkv + (size_t)BB * TT * NQKV;

  dim3 blk(128);

  qkv_gemm_kernel<<<dim3(NQKV / 64, (BB * TT) / 64), blk, 0, stream>>>(
      x, w_qkv, qkv, BB * TT, NQKV, CC);

  (void)hipFuncSetAttribute((const void*)flash_attn_kernel,
                            hipFuncAttributeMaxDynamicSharedMemorySize,
                            SM_TOTAL);
  flash_attn_kernel<<<dim3(TT / BQ, BB), blk, SM_TOTAL, stream>>>(qkv, o);

  out_gemm_kernel<<<dim3(CC / 64, (BB * TT) / 64), blk, 0, stream>>>(
      o, w_out, out, BB * TT, CC, CC);
}
